// BatchedMoE_18451179504158
// MI455X (gfx1250) — matrix operand order repaired and checked by reading, not compiled, not run on hardware
//
#include <hip/hip_runtime.h>
#include <hip/hip_bf16.h>

#define N_EXPERTS 8
#define D_MODEL   1024
#define D_FF      2048
#define N_TOKENS  4096
#define TOP_K     2
#define N_PAIRS   (N_TOKENS * TOP_K)

typedef __attribute__((ext_vector_type(16))) __bf16 v16bf;
typedef __attribute__((ext_vector_type(8)))  __bf16 v8bf;
typedef __attribute__((ext_vector_type(8)))  float  v8f;

__device__ __forceinline__ unsigned short f32_to_bf16(float f) {
  unsigned u = __float_as_uint(f);
  unsigned r = u + 0x7FFFu + ((u >> 16) & 1u);   // round to nearest even
  return (unsigned short)(r >> 16);
}

// A fragment (16x32 bf16, row-major source). Lane half h: K = h*16 + 0..15
__device__ __forceinline__ v16bf load_a_frag(const unsigned short* __restrict__ row, int k0, int h) {
  v8bf lo = *reinterpret_cast<const v8bf*>(row + k0 + h * 16);
  v8bf hi = *reinterpret_cast<const v8bf*>(row + k0 + h * 16 + 8);
  v16bf r;
#pragma unroll
  for (int i = 0; i < 8; ++i) { r[i] = lo[i]; r[i + 8] = hi[i]; }
  return r;
}

// B fragment (32x16 bf16). col = pointer to contiguous-K column; lane half h: K = h*16 + 0..15
__device__ __forceinline__ v16bf load_b_frag(const unsigned short* __restrict__ col, int k0, int h) {
  v8bf lo = *reinterpret_cast<const v8bf*>(col + k0 + h * 16);
  v8bf hi = *reinterpret_cast<const v8bf*>(col + k0 + h * 16 + 8);
  v16bf r;
#pragma unroll
  for (int i = 0; i < 8; ++i) { r[i] = lo[i]; r[i + 8] = hi[i]; }
  return r;
}

__device__ __forceinline__ v8f wmma_bf16(v16bf a, v16bf b, v8f c) {
  return __builtin_amdgcn_wmma_f32_16x16x32_bf16(false, a, false, b, (short)0, c, false, false);
}

// ---------------- conversion kernels ----------------

__global__ __launch_bounds__(256) void cvt_x_kernel(const float* __restrict__ in,
                                                    unsigned short* __restrict__ out) {
  int i = blockIdx.x * 256 + threadIdx.x;
  out[i] = f32_to_bf16(in[i]);
}

// in: [E][R][C] fp32 -> out: [E][C][R] bf16
__global__ __launch_bounds__(256) void transpose_cvt_kernel(const float* __restrict__ in,
                                                            unsigned short* __restrict__ out,
                                                            int R, int C) {
  __shared__ float tile[32][33];
  const int e = blockIdx.z;
  const float* inp = in + (size_t)e * R * C;
  unsigned short* outp = out + (size_t)e * R * C;
  const int c0 = blockIdx.x * 32, r0 = blockIdx.y * 32;
  const int tx = threadIdx.x, ty = threadIdx.y;   // 32 x 8
#pragma unroll
  for (int i = 0; i < 32; i += 8)
    tile[ty + i][tx] = inp[(size_t)(r0 + ty + i) * C + (c0 + tx)];
  __syncthreads();
#pragma unroll
  for (int i = 0; i < 32; i += 8)
    outp[(size_t)(c0 + ty + i) * R + (r0 + tx)] = f32_to_bf16(tile[tx][ty + i]);
}

// ---------------- dispatch kernels ----------------

__global__ void moe_count_kernel(const int* __restrict__ eidx, int* __restrict__ counts) {
  int p = blockIdx.x * 256 + threadIdx.x;
  if (p < N_PAIRS) atomicAdd(&counts[eidx[p]], 1);
}

__global__ void moe_scan_kernel(const int* __restrict__ counts, int* __restrict__ offsets,
                                int* __restrict__ running) {
  if (threadIdx.x == 0) {
    int acc = 0;
#pragma unroll
    for (int e = 0; e < N_EXPERTS; ++e) { offsets[e] = acc; running[e] = acc; acc += counts[e]; }
    offsets[N_EXPERTS] = acc;
  }
}

__global__ void moe_scatter_kernel(const int* __restrict__ eidx, const float* __restrict__ ew,
                                   int* __restrict__ running, int* __restrict__ tok_of_slot,
                                   float* __restrict__ wt_of_slot) {
  int p = blockIdx.x * 256 + threadIdx.x;
  if (p < N_PAIRS) {
    int e = eidx[p];
    int pos = atomicAdd(&running[e], 1);
    tok_of_slot[pos] = p >> 1;          // token id (p = t*TOP_K + k)
    wt_of_slot[pos]  = ew[p];
  }
}

// ---------------- GEMM1: hidden = silu(x@w1) * (x@w2), per selected pair ----------------
// grid: (N_PAIRS/64, E, F/128); block 256 = 8 waves as 4m x 2n; block tile 64x128,
// wave tile 16x64 with dual (gate,value) accumulators.

__global__ __launch_bounds__(256) void moe_gemm1_kernel(
    const unsigned short* __restrict__ xb,    // [T][D] bf16
    const unsigned short* __restrict__ w1t,   // [E][F][D] bf16
    const unsigned short* __restrict__ w2t,   // [E][F][D] bf16
    const int* __restrict__ tok_of_slot,
    const int* __restrict__ offsets,          // [E+1]
    unsigned short* __restrict__ hidden)      // [P][F] bf16
{
  const int e   = blockIdx.y;
  const int beg = offsets[e];
  const int cnt = offsets[e + 1] - beg;
  const int mbase = blockIdx.x * 64;
  if (mbase >= cnt) return;                    // uniform per block
  const int nblock = blockIdx.z * 128;

  const int lane = threadIdx.x & 31;
  const int wave = threadIdx.x >> 5;
  const int wm = wave >> 1;                    // 0..3
  const int wn = wave & 1;                     // 0..1
  const int h  = lane >> 4;
  const int ln = lane & 15;

  // gathered A row for this lane
  int mrow = mbase + wm * 16 + ln;
  int slot = beg + ((mrow < cnt) ? mrow : (cnt - 1));
  const unsigned short* arow = xb + (size_t)tok_of_slot[slot] * D_MODEL;

  const unsigned short* w1e = w1t + (size_t)e * D_FF * D_MODEL;
  const unsigned short* w2e = w2t + (size_t)e * D_FF * D_MODEL;
  const unsigned short* bg[4];
  const unsigned short* bv[4];
#pragma unroll
  for (int j = 0; j < 4; ++j) {
    int f = nblock + wn * 64 + j * 16 + ln;
    bg[j] = w1e + (size_t)f * D_MODEL;
    bv[j] = w2e + (size_t)f * D_MODEL;
  }

  v8f cg[4], cv[4];
#pragma unroll
  for (int j = 0; j < 4; ++j) { cg[j] = (v8f)0.0f; cv[j] = (v8f)0.0f; }

  for (int k0 = 0; k0 < D_MODEL; k0 += 32) {
    v16bf a = load_a_frag(arow, k0, h);
#pragma unroll
    for (int j = 0; j < 4; ++j) {
      v16bf b1 = load_b_frag(bg[j], k0, h);
      cg[j] = wmma_bf16(a, b1, cg[j]);
      v16bf b2 = load_b_frag(bv[j], k0, h);
      cv[j] = wmma_bf16(a, b2, cv[j]);
    }
  }

  // epilogue: SiLU(gate) * value -> bf16 hidden
#pragma unroll
  for (int j = 0; j < 4; ++j) {
    int f = nblock + wn * 64 + j * 16 + ln;
#pragma unroll
    for (int r = 0; r < 8; ++r) {
      int m = mbase + wm * 16 + 8 * h + r;
      if (m < cnt) {
        float g = cg[j][r];
        float hv = (g / (1.0f + __expf(-g))) * cv[j][r];
        hidden[(size_t)(beg + m) * D_FF + f] = f32_to_bf16(hv);
      }
    }
  }
}

// ---------------- GEMM2: out[t] += wt * hidden[slot] @ w3[e] ----------------
// grid: (N_PAIRS/128, E, D/128); block 256 = 8 waves as 8m x 1n; block tile 128x128,
// wave tile 16x128 (8 accumulators).

__global__ __launch_bounds__(256) void moe_gemm2_kernel(
    const unsigned short* __restrict__ hidden,  // [P][F] bf16
    const unsigned short* __restrict__ w3t,     // [E][D][F] bf16
    const int* __restrict__ tok_of_slot,
    const float* __restrict__ wt_of_slot,
    const int* __restrict__ offsets,
    float* __restrict__ out)                    // [T][D] fp32 (pre-zeroed)
{
  const int e   = blockIdx.y;
  const int beg = offsets[e];
  const int cnt = offsets[e + 1] - beg;
  const int mbase = blockIdx.x * 128;
  if (mbase >= cnt) return;
  const int nblock = blockIdx.z * 128;

  const int lane = threadIdx.x & 31;
  const int wm   = threadIdx.x >> 5;           // 0..7 (8 m-subtiles)
  const int h  = lane >> 4;
  const int ln = lane & 15;

  int mrow = mbase + wm * 16 + ln;
  int slotA = beg + ((mrow < cnt) ? mrow : (cnt - 1));
  const unsigned short* arow = hidden + (size_t)slotA * D_FF;

  const unsigned short* w3e = w3t + (size_t)e * D_MODEL * D_FF;
  const unsigned short* bc[8];
#pragma unroll
  for (int j = 0; j < 8; ++j) {
    int d = nblock + j * 16 + ln;
    bc[j] = w3e + (size_t)d * D_FF;
  }

  v8f c[8];
#pragma unroll
  for (int j = 0; j < 8; ++j) c[j] = (v8f)0.0f;

  for (int k0 = 0; k0 < D_FF; k0 += 32) {
    v16bf a = load_a_frag(arow, k0, h);
#pragma unroll
    for (int j = 0; j < 8; ++j) {
      v16bf b = load_b_frag(bc[j], k0, h);
      c[j] = wmma_bf16(a, b, c[j]);
    }
  }

#pragma unroll
  for (int r = 0; r < 8; ++r) {
    int m = mbase + wm * 16 + 8 * h + r;
    if (m < cnt) {
      int s = beg + m;
      int t = tok_of_slot[s];
      float w = wt_of_slot[s];
#pragma unroll
      for (int j = 0; j < 8; ++j) {
        int d = nblock + j * 16 + ln;
        atomicAdd(&out[(size_t)t * D_MODEL + d], w * c[j][r]);
      }
    }
  }
}

// ---------------- host launch ----------------

extern "C" void kernel_launch(void* const* d_in, const int* in_sizes, int n_in,
                              void* d_out, int out_size, void* d_ws, size_t ws_size,
                              hipStream_t stream) {
  const float* x    = (const float*)d_in[0];
  const int*   eidx = (const int*)  d_in[1];
  const float* ew   = (const float*)d_in[2];
  const float* w1   = (const float*)d_in[3];
  const float* w2   = (const float*)d_in[4];
  const float* w3   = (const float*)d_in[5];
  float* out = (float*)d_out;

  unsigned char* base = (unsigned char*)d_ws;
  size_t off = 0;
  auto take = [&](size_t bytes) -> void* {
    void* p = base + off;
    off += (bytes + 255) & ~(size_t)255;
    return p;
  };

  unsigned short* xb     = (unsigned short*)take((size_t)N_TOKENS * D_MODEL * 2);
  unsigned short* w1t    = (unsigned short*)take((size_t)N_EXPERTS * D_MODEL * D_FF * 2);
  unsigned short* w2t    = (unsigned short*)take((size_t)N_EXPERTS * D_MODEL * D_FF * 2);
  unsigned short* w3t    = (unsigned short*)take((size_t)N_EXPERTS * D_MODEL * D_FF * 2);
  unsigned short* hidden = (unsigned short*)take((size_t)N_PAIRS * D_FF * 2);
  int*   tok_of_slot = (int*)  take((size_t)N_PAIRS * 4);
  float* wt_of_slot  = (float*)take((size_t)N_PAIRS * 4);
  int*   counts      = (int*)  take(N_EXPERTS * 4);
  int*   offsets     = (int*)  take((N_EXPERTS + 1) * 4);
  int*   running     = (int*)  take(N_EXPERTS * 4);

  hipMemsetAsync(counts, 0, N_EXPERTS * sizeof(int), stream);
  hipMemsetAsync(out, 0, (size_t)N_TOKENS * D_MODEL * sizeof(float), stream);

  // bf16 staging (weights become L2-resident: 96 MB bf16 << 192 MB L2)
  cvt_x_kernel<<<(N_TOKENS * D_MODEL) / 256, 256, 0, stream>>>(x, xb);
  transpose_cvt_kernel<<<dim3(D_FF / 32, D_MODEL / 32, N_EXPERTS), dim3(32, 8), 0, stream>>>(
      w1, w1t, D_MODEL, D_FF);
  transpose_cvt_kernel<<<dim3(D_FF / 32, D_MODEL / 32, N_EXPERTS), dim3(32, 8), 0, stream>>>(
      w2, w2t, D_MODEL, D_FF);
  transpose_cvt_kernel<<<dim3(D_MODEL / 32, D_FF / 32, N_EXPERTS), dim3(32, 8), 0, stream>>>(
      w3, w3t, D_FF, D_MODEL);

  // top-k dispatch
  moe_count_kernel<<<N_PAIRS / 256, 256, 0, stream>>>(eidx, counts);
  moe_scan_kernel<<<1, 32, 0, stream>>>(counts, offsets, running);
  moe_scatter_kernel<<<N_PAIRS / 256, 256, 0, stream>>>(eidx, ew, running, tok_of_slot, wt_of_slot);

  // WMMA GEMMs (grid.x sized for the worst-case expert load; empty tiles exit)
  moe_gemm1_kernel<<<dim3(N_PAIRS / 64, N_EXPERTS, D_FF / 128), 256, 0, stream>>>(
      xb, w1t, w2t, tok_of_slot, offsets, hidden);
  moe_gemm2_kernel<<<dim3(N_PAIRS / 128, N_EXPERTS, D_MODEL / 128), 256, 0, stream>>>(
      hidden, w3t, tok_of_slot, wt_of_slot, offsets, out);
}
